// DiffAttn_83803401879657
// MI455X (gfx1250) — matrix operand order repaired and checked by reading, not compiled, not run on hardware
//
#include <hip/hip_runtime.h>
#include <hip/hip_fp16.h>
#include <math.h>

typedef __attribute__((ext_vector_type(16))) _Float16 v16h;
typedef __attribute__((ext_vector_type(8)))  _Float16 v8h;
typedef __attribute__((ext_vector_type(4)))  _Float16 v4h;
typedef __attribute__((ext_vector_type(8)))  float    v8f;
typedef __attribute__((ext_vector_type(4)))  unsigned int u32x4;
typedef __attribute__((ext_vector_type(8)))  int      i32x8;
typedef __attribute__((ext_vector_type(4)))  int      i32x4;

#define WMMA_F16(a, b, c) \
  __builtin_amdgcn_wmma_f32_16x16x32_f16(false, (a), false, (b), (short)0, (c), false, false)

#if defined(__gfx1250__) && __has_builtin(__builtin_amdgcn_tensor_load_to_lds)
#define HAVE_TDM 1
#else
#define HAVE_TDM 0
#endif

constexpr int   B_      = 2;
constexpr int   T_      = 2048;
constexpr int   HN      = 8;      // heads for v / output
constexpr int   DH      = 32;     // dim head for q/k
constexpr int   IN_DIM  = 1024;
constexpr int   E_      = 512;    // 2*H*D
constexpr int   OUT_DIM = 1024;
constexpr float LAMBDA_INIT = 0.3555090675909693f;   // 0.8 - 0.6*exp(-0.3)
constexpr float QK_SCALE    = 0.17677669529663687f;  // D^-0.5

// ---------------- TDM: 2D tile (f16 elements) global -> LDS -----------------
// D# per CDNA5 ISA 8.3/8.4: group0 = {count|flags, lds_addr, global_addr, type},
// group1 = {mask|data_size, tensor dims, tile dims, dim0 stride}.
__device__ __forceinline__ void tdm_load_tile(unsigned lds_off, const void* gptr,
                                              unsigned tile0, unsigned tile1,
                                              unsigned tdim0, unsigned tdim1,
                                              unsigned stride0) {
#if HAVE_TDM
  const unsigned long long ga = (unsigned long long)gptr;
  u32x4 g0;
  g0[0] = 1u;                                    // count=1, user descriptor
  g0[1] = lds_off;                               // lds_addr (bytes)
  g0[2] = (unsigned)ga;                          // global_addr[31:0]
  g0[3] = (unsigned)((ga >> 32) & 0x01ffffffu) | (2u << 30);  // addr[56:32] | type=2
  i32x8 g1;
  g1[0] = (int)(1u << 16);                       // workgroup_mask=0, data_size=1 (2B)
  g1[1] = (int)((tdim0 & 0xffffu) << 16);        // tensor_dim0[15:0]
  g1[2] = (int)(((tdim0 >> 16) & 0xffffu) | ((tdim1 & 0xffffu) << 16));
  g1[3] = (int)(((tdim1 >> 16) & 0xffffu) | ((tile0 & 0xffffu) << 16));
  g1[4] = (int)(tile1 & 0xffffu);                // tile_dim1 (0 = 1D), tile_dim2=0
  g1[5] = (int)stride0;                          // tensor_dim0_stride[31:0]
  g1[6] = 0;
  g1[7] = 0;
  i32x4 z4 = {0, 0, 0, 0};
#if __clang_major__ >= 23
  i32x8 z8 = {0, 0, 0, 0, 0, 0, 0, 0};
  __builtin_amdgcn_tensor_load_to_lds(g0, g1, z4, z4, z8, 0);
#else
  __builtin_amdgcn_tensor_load_to_lds(g0, g1, z4, z4, 0);
#endif
#else
  (void)lds_off; (void)gptr; (void)tile0; (void)tile1;
  (void)tdim0; (void)tdim1; (void)stride0;
#endif
}

// ---------------- WMMA fragment loaders (vectorized, b128 LDS reads) --------
__device__ __forceinline__ v16h cat8(v8h lo, v8h hi) {
  return __builtin_shufflevector(lo, hi, 0, 1, 2, 3, 4, 5, 6, 7,
                                 8, 9, 10, 11, 12, 13, 14, 15);
}
// A matrix 16(M) x 32(K) f16, row-major source:
// lane -> M=lane&15 ; elements = k in [kb,kb+8) ++ [kb+16,kb+24), kb=(lane&16)?8:0
__device__ __forceinline__ v16h load_a_frag(const _Float16* __restrict__ src, int ld) {
  const int lane = threadIdx.x & 31;
  const int r  = lane & 15;
  const int kb = (lane & 16) ? 8 : 0;
  const v8h lo = *(const v8h*)(src + r * ld + kb);
  const v8h hi = *(const v8h*)(src + r * ld + kb + 16);
  return cat8(lo, hi);
}
// B matrix 32(K) x 16(N) f16 from N-major storage: element(k,n) = src[n*ld + k].
__device__ __forceinline__ v16h load_b_frag_nmaj(const _Float16* __restrict__ src, int ld) {
  const int lane = threadIdx.x & 31;
  const int n  = lane & 15;
  const int kb = (lane & 16) ? 8 : 0;
  const v8h lo = *(const v8h*)(src + n * ld + kb);
  const v8h hi = *(const v8h*)(src + n * ld + kb + 16);
  return cat8(lo, hi);
}

// ---------------- kernel 0: lambda scalar -----------------------------------
__global__ void lam_kernel(const float* __restrict__ lq1, const float* __restrict__ lk1,
                           const float* __restrict__ lq2, const float* __restrict__ lk2,
                           float* __restrict__ lam_out) {
  const int lane = threadIdx.x & 31;   // D == 32 == wave32
  float s1 = lq1[lane] * lk1[lane];
  float s2 = lq2[lane] * lk2[lane];
#pragma unroll
  for (int m = 16; m >= 1; m >>= 1) {
    s1 += __shfl_xor(s1, m, 32);
    s2 += __shfl_xor(s2, m, 32);
  }
  if (lane == 0) *lam_out = __expf(s1) - __expf(s2) + LAMBDA_INIT;
}

// ---------------- prep A: elementwise f32 -> f16 (for x) --------------------
__global__ __launch_bounds__(128)
void cvt_f32_f16(const float* __restrict__ src, _Float16* __restrict__ dst) {
  const size_t i = ((size_t)blockIdx.x * 128 + threadIdx.x) * 8;
  const float4 a = *(const float4*)(src + i);
  const float4 b = *(const float4*)(src + i + 4);
  v8h o;
  o[0] = (_Float16)a.x; o[1] = (_Float16)a.y; o[2] = (_Float16)a.z; o[3] = (_Float16)a.w;
  o[4] = (_Float16)b.x; o[5] = (_Float16)b.y; o[6] = (_Float16)b.z; o[7] = (_Float16)b.w;
  *(v8h*)(dst + i) = o;
}

// ---------------- prep B: transpose + convert weights to N-major f16 --------
// src: f32 [K][N] row-major.  dst: f16 [N][K] row-major.
__global__ __launch_bounds__(128)
void transpose_cvt(const float* __restrict__ src, _Float16* __restrict__ dst,
                   int K, int N) {
  __shared__ __align__(16) _Float16 t[64][32];   // [n][k] tile
  const int tid = threadIdx.x;
  const int kt  = blockIdx.x % (K / 32);
  const int nt  = blockIdx.x / (K / 32);
  const int r   = tid >> 2;          // k-local 0..31
  const int c0  = (tid & 3) * 16;    // n-local base
#pragma unroll
  for (int j = 0; j < 4; ++j) {      // coalesced f32 reads along n
    const float4 f =
        *(const float4*)&src[(size_t)(kt * 32 + r) * N + nt * 64 + c0 + 4 * j];
    t[c0 + 4 * j + 0][r] = (_Float16)f.x;
    t[c0 + 4 * j + 1][r] = (_Float16)f.y;
    t[c0 + 4 * j + 2][r] = (_Float16)f.z;
    t[c0 + 4 * j + 3][r] = (_Float16)f.w;
  }
  __syncthreads();
  const int n  = tid >> 1;           // n-local 0..63
  const int ko = (tid & 1) * 16;     // k-local base
  _Float16* d = &dst[(size_t)(nt * 64 + n) * K + kt * 32 + ko];
  *(v8h*)d       = *(const v8h*)&t[n][ko];
  *(v8h*)(d + 8) = *(const v8h*)&t[n][ko + 8];
}

// ---------------- kernel 1: fused QKV projection GEMM (TDM pipelined) -------
// Y[4096,1536] = xh[4096,1024] @ Wt[1536,1024]^T ; scatter q,k -> [B,16,T,32],
// v -> [B,H,2D,T] (transposed so attention stages V with no LDS transpose).
__global__ __launch_bounds__(128)
void qkv_gemm(const _Float16* __restrict__ xh, const _Float16* __restrict__ Wt,
              _Float16* __restrict__ qo, _Float16* __restrict__ ko,
              _Float16* __restrict__ vo) {
  __shared__ __align__(128) _Float16 As[2][64 * 32];   // [m][k] row-major
  __shared__ __align__(128) _Float16 Bs[2][64 * 32];   // [n][k] N-major
  const int tid  = threadIdx.x;
  const int lane = tid & 31;
  const int wave = tid >> 5;
  const int m0 = (blockIdx.x & 63) * 64;   // 64 M tiles
  const int n0 = (blockIdx.x >> 6) * 64;   // 24 N tiles
  const _Float16* abase = xh + (size_t)m0 * IN_DIM;
  const _Float16* bbase = Wt + (size_t)n0 * IN_DIM;
  const unsigned as_off[2] = {(unsigned)(unsigned long long)&As[0][0],
                              (unsigned)(unsigned long long)&As[1][0]};
  const unsigned bs_off[2] = {(unsigned)(unsigned long long)&Bs[0][0],
                              (unsigned)(unsigned long long)&Bs[1][0]};

  v8f acc[4] = {};
  int p = 0;
#if HAVE_TDM
  if (wave == 0) {
    tdm_load_tile(as_off[0], abase, 32, 64, 32, 64, IN_DIM);
    tdm_load_tile(bs_off[0], bbase, 32, 64, 32, 64, IN_DIM);
  }
#endif
  for (int k0 = 0; k0 < IN_DIM; k0 += 32) {
#if HAVE_TDM
    if (k0 + 32 < IN_DIM) {
      if (wave == 0) {
        tdm_load_tile(as_off[p ^ 1], abase + k0 + 32, 32, 64, 32, 64, IN_DIM);
        tdm_load_tile(bs_off[p ^ 1], bbase + k0 + 32, 32, 64, 32, 64, IN_DIM);
      }
      __builtin_amdgcn_s_wait_tensorcnt(2);
    } else {
      __builtin_amdgcn_s_wait_tensorcnt(0);
    }
    __syncthreads();
#else
    __syncthreads();
    {
      const int r = tid >> 1, ko2 = (tid & 1) * 16;
      *(v8h*)&As[p][r * 32 + ko2]     = *(const v8h*)(abase + (size_t)r * IN_DIM + k0 + ko2);
      *(v8h*)&As[p][r * 32 + ko2 + 8] = *(const v8h*)(abase + (size_t)r * IN_DIM + k0 + ko2 + 8);
      *(v8h*)&Bs[p][r * 32 + ko2]     = *(const v8h*)(bbase + (size_t)r * IN_DIM + k0 + ko2);
      *(v8h*)&Bs[p][r * 32 + ko2 + 8] = *(const v8h*)(bbase + (size_t)r * IN_DIM + k0 + ko2 + 8);
    }
    __syncthreads();
#endif
    const v16h a = load_a_frag(&As[p][wave * 16 * 32], 32);
#pragma unroll
    for (int n = 0; n < 4; ++n) {
      const v16h bq = load_b_frag_nmaj(&Bs[p][(n * 16) * 32], 32);
      acc[n] = WMMA_F16(a, bq, acc[n]);
    }
#if HAVE_TDM
    __syncthreads();
    p ^= 1;
#endif
  }
  // scatter C fragments
  const int col_l = lane & 15;
  const int rbase = (lane & 16) ? 8 : 0;
#pragma unroll
  for (int n = 0; n < 4; ++n) {
#pragma unroll
    for (int r = 0; r < 8; ++r) {
      const int m = m0 + wave * 16 + rbase + r;
      const int ncol = n0 + n * 16 + col_l;
      const int b = m / T_, t = m % T_;
      const float vv = acc[n][r];
      if (ncol < E_) {                         // q -> [B,16,T,32], pre-scaled
        const int h2 = ncol / DH, d = ncol % DH;
        qo[(((size_t)(b * 16 + h2) * T_) + t) * DH + d] = (_Float16)(vv * QK_SCALE);
      } else if (ncol < 2 * E_) {              // k -> [B,16,T,32]
        const int j = ncol - E_;
        const int h2 = j / DH, d = j % DH;
        ko[(((size_t)(b * 16 + h2) * T_) + t) * DH + d] = (_Float16)vv;
      } else {                                 // v -> [B,H,2D,T] transposed
        const int j = ncol - 2 * E_;
        const int h = j / 64, d = j % 64;
        vo[((size_t)((b * HN + h) * 64 + d)) * T_ + t] = (_Float16)vv;
      }
    }
  }
}

// ---------------- kernel 2: streaming differential flash attention ----------
// K/V chunks delivered by the Tensor Data Mover (double buffered, TENSORcnt).
__global__ __launch_bounds__(128)
void diff_attn(const _Float16* __restrict__ q, const _Float16* __restrict__ k,
               const _Float16* __restrict__ v, const float* __restrict__ gamma,
               const float* __restrict__ lam_p, _Float16* __restrict__ ao) {
  __shared__ __align__(128) _Float16 Ks[2][64 * 32];    // [key][d]  (N-major for QK^T)
  __shared__ __align__(128) _Float16 Vs[2][64 * 64];    // [d][key]  (N-major for P.V)
  __shared__ __align__(16)  _Float16 Ps[4][16 * 64];    // per-wave P, row-major

  const int tid  = threadIdx.x;
  const int lane = tid & 31;
  const int wave = tid >> 5;
  const int qt = blockIdx.x & 31;                  // T/64 q tiles
  const int h  = (blockIdx.x >> 5) & 7;
  const int b  = blockIdx.x >> 8;
  const int t0 = qt * 64 + wave * 16;              // this wave's 16 q rows
  const int col_l = lane & 15;
  const int rbase = (lane & 16) ? 8 : 0;
  const float lam = *lam_p;

  const unsigned ks_off[2] = {(unsigned)(unsigned long long)&Ks[0][0],
                              (unsigned)(unsigned long long)&Ks[1][0]};
  const unsigned vs_off[2] = {(unsigned)(unsigned long long)&Vs[0][0],
                              (unsigned)(unsigned long long)&Vs[1][0]};
#if !HAVE_TDM
  const int dV    = tid >> 1;                      // fallback staging assignment
  const int kVoff = (tid & 1) * 32;
#endif

  v8f Oc[2][4];
  for (int c = 0; c < 2; ++c) {
    const int hc = 2 * h + c;                      // component head
    const _Float16* khead = k + ((size_t)(b * 16 + hc) * T_) * DH;
    const _Float16* vhead = v + ((size_t)(b * HN + h) * 64) * T_;
    const v16h qa = load_a_frag(q + (((size_t)(b * 16 + hc) * T_) + t0) * DH, DH);
    v8f O[4] = {};
    float l8[8], m8[8];
#pragma unroll
    for (int r = 0; r < 8; ++r) { l8[r] = 0.0f; m8[r] = -1e30f; }

    int p = 0;
#if HAVE_TDM
    if (wave == 0) {   // prologue: chunk 0 (K: 1D 4KB tile, V: 2D 64x64 tile)
      tdm_load_tile(ks_off[0], khead, 2048, 0, 2048, 1, 2048);
      tdm_load_tile(vs_off[0], vhead, 64, 64, 64, 64, T_);
    }
#endif
    for (int kk0 = 0; kk0 < T_; kk0 += 64) {
#if HAVE_TDM
      if (kk0 + 64 < T_) {                         // issue chunk i+1, wait chunk i
        if (wave == 0) {
          tdm_load_tile(ks_off[p ^ 1], khead + (size_t)(kk0 + 64) * DH,
                        2048, 0, 2048, 1, 2048);
          tdm_load_tile(vs_off[p ^ 1], vhead + (kk0 + 64), 64, 64, 64, 64, T_);
        }
        __builtin_amdgcn_s_wait_tensorcnt(2);
      } else {
        __builtin_amdgcn_s_wait_tensorcnt(0);
      }
      __syncthreads();
#else
      __syncthreads();
      {  // fallback: vectorized cooperative staging
        const _Float16* kbase = khead + (size_t)kk0 * DH;
        const int off = tid * 16;
        *(v8h*)&Ks[p][off]     = *(const v8h*)(kbase + off);
        *(v8h*)&Ks[p][off + 8] = *(const v8h*)(kbase + off + 8);
        const _Float16* vbase = vhead + (size_t)dV * T_ + kk0 + kVoff;
#pragma unroll
        for (int u = 0; u < 4; ++u)
          *(v8h*)&Vs[p][dV * 64 + kVoff + u * 8] = *(const v8h*)(vbase + u * 8);
      }
      __syncthreads();
#endif

      // S = Q . K^T   (16 x 64 chunk, 4 wmma; K-dim = D = 32 exactly)
      v8f S[4] = {};
#pragma unroll
      for (int n = 0; n < 4; ++n) {
        const v16h bk = load_b_frag_nmaj(&Ks[p][(n * 16) * 32], 32);
        S[n] = WMMA_F16(qa, bk, S[n]);
      }

      // online softmax: row stats per VGPR index, reduced across 16-lane groups
      float alpha[8];
#pragma unroll
      for (int r = 0; r < 8; ++r) {
        float cm = S[0][r];
#pragma unroll
        for (int n = 1; n < 4; ++n) cm = fmaxf(cm, S[n][r]);
#pragma unroll
        for (int m = 8; m >= 1; m >>= 1) cm = fmaxf(cm, __shfl_xor(cm, m, 32));
        const float nm = fmaxf(m8[r], cm);
        alpha[r] = __expf(m8[r] - nm);
        m8[r] = nm;
      }
      float rs[8];
#pragma unroll
      for (int r = 0; r < 8; ++r) rs[r] = 0.0f;
#pragma unroll
      for (int n = 0; n < 4; ++n) {
#pragma unroll
        for (int r = 0; r < 8; ++r) {
          const float pe = __expf(S[n][r] - m8[r]);
          S[n][r] = pe;
          rs[r] += pe;
        }
      }
#pragma unroll
      for (int r = 0; r < 8; ++r) {
#pragma unroll
        for (int m = 8; m >= 1; m >>= 1) rs[r] += __shfl_xor(rs[r], m, 32);
        l8[r] = l8[r] * alpha[r] + rs[r];
      }
#pragma unroll
      for (int n = 0; n < 4; ++n)
#pragma unroll
        for (int r = 0; r < 8; ++r) O[n][r] = O[n][r] * alpha[r];

      // stage P through per-wave LDS (C-layout -> A-layout; intra-wave DS in order)
#pragma unroll
      for (int n = 0; n < 4; ++n)
#pragma unroll
        for (int r = 0; r < 8; ++r)
          Ps[wave][(rbase + r) * 64 + n * 16 + col_l] = (_Float16)S[n][r];

      // O += P . V   (K-dim 64 -> 2 wmma halves x 4 dim subtiles)
#pragma unroll
      for (int kh = 0; kh < 2; ++kh) {
        const v16h pa = load_a_frag(&Ps[wave][kh * 32], 64);
#pragma unroll
        for (int n = 0; n < 4; ++n) {
          const v16h vb = load_b_frag_nmaj(&Vs[p][(n * 16) * 64 + kh * 32], 64);
          O[n] = WMMA_F16(pa, vb, O[n]);
        }
      }
#if HAVE_TDM
      __syncthreads();   // all waves done with buffer p before TDM overwrites it
      p ^= 1;
#endif
    }
    // normalize component
#pragma unroll
    for (int n = 0; n < 4; ++n)
#pragma unroll
      for (int r = 0; r < 8; ++r) Oc[c][n][r] = O[n][r] / l8[r];
  }

  // combine, RMS-norm over 64 dims, gamma, (1-lambda_init), store f16 [B,T,E]
  float g4[4];
#pragma unroll
  for (int n = 0; n < 4; ++n) g4[n] = gamma[n * 16 + col_l];
#pragma unroll
  for (int r = 0; r < 8; ++r) {
    float od[4];
    float ss = 0.0f;
#pragma unroll
    for (int n = 0; n < 4; ++n) {
      od[n] = Oc[0][n][r] - lam * Oc[1][n][r];
      ss += od[n] * od[n];
    }
#pragma unroll
    for (int m = 8; m >= 1; m >>= 1) ss += __shfl_xor(ss, m, 32);
    const float rms = sqrtf(ss * (1.0f / 64.0f));
    const float sc = (1.0f - LAMBDA_INIT) / (rms + 1e-8f);
    const int trow = t0 + rbase + r;
#pragma unroll
    for (int n = 0; n < 4; ++n)
      ao[((size_t)(b * T_ + trow)) * E_ + h * 64 + n * 16 + col_l] =
          (_Float16)(od[n] * sc * g4[n]);
  }
}

// ---------------- kernel 3: output projection GEMM (TDM pipelined) ----------
__global__ __launch_bounds__(128)
void out_gemm(const _Float16* __restrict__ a, const _Float16* __restrict__ Wt,
              float* __restrict__ out) {
  __shared__ __align__(128) _Float16 As[2][64 * 32];   // [m][k]
  __shared__ __align__(128) _Float16 Bs[2][64 * 32];   // [n][k] N-major
  const int tid  = threadIdx.x;
  const int lane = tid & 31;
  const int wave = tid >> 5;
  const int m0 = (blockIdx.x & 63) * 64;   // 64 M tiles
  const int n0 = (blockIdx.x >> 6) * 64;   // 16 N tiles
  const _Float16* abase = a + (size_t)m0 * E_;
  const _Float16* bbase = Wt + (size_t)n0 * E_;
  const unsigned as_off[2] = {(unsigned)(unsigned long long)&As[0][0],
                              (unsigned)(unsigned long long)&As[1][0]};
  const unsigned bs_off[2] = {(unsigned)(unsigned long long)&Bs[0][0],
                              (unsigned)(unsigned long long)&Bs[1][0]};

  v8f acc[4] = {};
  int p = 0;
#if HAVE_TDM
  if (wave == 0) {
    tdm_load_tile(as_off[0], abase, 32, 64, 32, 64, E_);
    tdm_load_tile(bs_off[0], bbase, 32, 64, 32, 64, E_);
  }
#endif
  for (int k0 = 0; k0 < E_; k0 += 32) {
#if HAVE_TDM
    if (k0 + 32 < E_) {
      if (wave == 0) {
        tdm_load_tile(as_off[p ^ 1], abase + k0 + 32, 32, 64, 32, 64, E_);
        tdm_load_tile(bs_off[p ^ 1], bbase + k0 + 32, 32, 64, 32, 64, E_);
      }
      __builtin_amdgcn_s_wait_tensorcnt(2);
    } else {
      __builtin_amdgcn_s_wait_tensorcnt(0);
    }
    __syncthreads();
#else
    __syncthreads();
    {
      const int r = tid >> 1, ko2 = (tid & 1) * 16;
      *(v8h*)&As[p][r * 32 + ko2]     = *(const v8h*)(abase + (size_t)r * E_ + k0 + ko2);
      *(v8h*)&As[p][r * 32 + ko2 + 8] = *(const v8h*)(abase + (size_t)r * E_ + k0 + ko2 + 8);
      *(v8h*)&Bs[p][r * 32 + ko2]     = *(const v8h*)(bbase + (size_t)r * E_ + k0 + ko2);
      *(v8h*)&Bs[p][r * 32 + ko2 + 8] = *(const v8h*)(bbase + (size_t)r * E_ + k0 + ko2 + 8);
    }
    __syncthreads();
#endif
    const v16h af = load_a_frag(&As[p][wave * 16 * 32], 32);
#pragma unroll
    for (int n = 0; n < 4; ++n) {
      const v16h bf = load_b_frag_nmaj(&Bs[p][(n * 16) * 32], 32);
      acc[n] = WMMA_F16(af, bf, acc[n]);
    }
#if HAVE_TDM
    __syncthreads();
    p ^= 1;
#endif
  }
  const int col_l = lane & 15;
  const int rbase = (lane & 16) ? 8 : 0;
#pragma unroll
  for (int n = 0; n < 4; ++n)
#pragma unroll
    for (int r = 0; r < 8; ++r)
      out[(size_t)(m0 + wave * 16 + rbase + r) * OUT_DIM + n0 + n * 16 + col_l] =
          acc[n][r];
}

// ---------------- launcher ---------------------------------------------------
extern "C" void kernel_launch(void* const* d_in, const int* in_sizes, int n_in,
                              void* d_out, int out_size, void* d_ws, size_t ws_size,
                              hipStream_t stream) {
  (void)in_sizes; (void)n_in; (void)out_size; (void)ws_size;
  const float* x    = (const float*)d_in[0];
  const float* Wq   = (const float*)d_in[1];
  const float* Wkv  = (const float*)d_in[2];
  const float* Wout = (const float*)d_in[3];
  const float* lq1  = (const float*)d_in[4];
  const float* lk1  = (const float*)d_in[5];
  const float* lq2  = (const float*)d_in[6];
  const float* lk2  = (const float*)d_in[7];
  const float* gam  = (const float*)d_in[8];

  char* ws = (char*)d_ws;
  float*     lam = (float*)ws;
  constexpr size_t QKV_ELEMS = (size_t)B_ * 16 * T_ * DH;   // 2,097,152 per buffer
  _Float16* qw  = (_Float16*)(ws + 256);
  _Float16* kw  = qw + QKV_ELEMS;
  _Float16* vw  = kw + QKV_ELEMS;                 // [B,H,2D,T]
  _Float16* aw  = vw + QKV_ELEMS;                 // [B,T,512]
  _Float16* xh  = aw + QKV_ELEMS;                 // [4096,1024] f16
  _Float16* wt1 = xh + (size_t)B_ * T_ * IN_DIM;  // [1536,1024] f16 (Wq|Wkv)^T
  _Float16* wt2 = wt1 + (size_t)(3 * E_) * IN_DIM;// [1024,512]  f16 Wout^T

  lam_kernel<<<1, 32, 0, stream>>>(lq1, lk1, lq2, lk2, lam);
  cvt_f32_f16<<<(B_ * T_ * IN_DIM) / (128 * 8), 128, 0, stream>>>(x, xh);
  transpose_cvt<<<(IN_DIM / 32) * (E_ / 64),      128, 0, stream>>>(Wq,  wt1, IN_DIM, E_);
  transpose_cvt<<<(IN_DIM / 32) * (2 * E_ / 64),  128, 0, stream>>>(
      Wkv, wt1 + (size_t)E_ * IN_DIM, IN_DIM, 2 * E_);
  transpose_cvt<<<(E_ / 32) * (OUT_DIM / 64),     128, 0, stream>>>(Wout, wt2, E_, OUT_DIM);

  qkv_gemm<<<64 * 24, 128, 0, stream>>>(xh, wt1, qw, kw, vw);
  diff_attn<<<B_ * HN * (T_ / 64), 128, 0, stream>>>(qw, kw, vw, gam, lam, aw);
  out_gemm<<<64 * 16, 128, 0, stream>>>(aw, wt2, (float*)d_out);
}
